// SimilarityTreeLSTM_88201448390792
// MI455X (gfx1250) — compile-verified, matrix operand order checked by reading, not run on hardware
//
#include <hip/hip_runtime.h>
#include <hip/hip_bf16.h>
#include <math.h>

// ---------------------------------------------------------------------------
// SimilarityTreeLSTM for MI455X (gfx1250, wave32, WMMA).
// Strategy: hoist token-dependent projections into batched f16 WMMA GEMMs,
// keep the sequential 128-step tree scan minimal (1 workgroup/tree, LDS gate
// scratch), WMMA again for the GRU input projections, tiny VALU head.
// ---------------------------------------------------------------------------

typedef __attribute__((ext_vector_type(16))) _Float16 v16h;
typedef __attribute__((ext_vector_type(8)))  _Float16 v8h;
typedef __attribute__((ext_vector_type(8)))  float    v8f;

#define NNODE 128
#define INDIM 300
#define INP   320      // INDIM padded to 32
#define MEMD  150
#define IOU   450
#define IOUP  464      // 450 padded to 16
#define FXP   160      // 150 padded to 16
#define GH3   1350
#define GH3P  1360     // 1350 padded to 16
#define CONK  450
#define CONKP 480      // 450 padded to 32
#define CHS   152      // C/H row stride (floats)

__device__ __forceinline__ float sigf(float x) { return 1.0f / (1.0f + __expf(-x)); }

// ---------------------------------------------------------------------------
// f32 -> f16 conversion with zero padding: src [M x K] -> dst [Mp x Kp]
// ---------------------------------------------------------------------------
__global__ void k_cvt_pad(const float* __restrict__ src, _Float16* __restrict__ dst,
                          int M, int K, int Mp, int Kp) {
  int idx = blockIdx.x * blockDim.x + threadIdx.x;
  if (idx >= Mp * Kp) return;
  int r = idx / Kp, c = idx % Kp;
  float v = (r < M && c < K) ? src[(size_t)r * K + c] : 0.0f;
  dst[idx] = (_Float16)v;
}

// ---------------------------------------------------------------------------
// Embedding gather for both trees -> f16 padded X [2][128][INP]
// ---------------------------------------------------------------------------
__global__ void k_gather(const int* __restrict__ ltok, const int* __restrict__ rtok,
                         const float* __restrict__ emb, _Float16* __restrict__ X16) {
  int idx = blockIdx.x * blockDim.x + threadIdx.x;
  if (idx >= 2 * NNODE * INP) return;
  int tree = idx / (NNODE * INP);
  int rem  = idx % (NNODE * INP);
  int node = rem / INP;
  int k    = rem % INP;
  const int* tok = tree ? rtok : ltok;
  float v = 0.0f;
  if (k < INDIM) v = emb[(size_t)tok[node] * INDIM + k];
  X16[idx] = (_Float16)v;
}

// ---------------------------------------------------------------------------
// WMMA tile loader: 16x32 f16 tile in the ISA 7.12.2 A/B register layout from
// a row-major [rows x ld] matrix (K contiguous). Lane L%16 = row, lane/16
// selects the K chunk; two contiguous 16B loads per lane (global_load_b128).
// Caller guarantees padded dims (no guards needed).
// ---------------------------------------------------------------------------
__device__ __forceinline__ v16h load_tile16x32(const _Float16* base, int ld,
                                               int row0, int k0, int lane) {
  int r     = row0 + (lane & 15);
  int chunk = lane >> 4;                 // 0 or 1
  const _Float16* p0 = base + (size_t)r * ld + k0 + 8 * chunk; // K = k0+{0,8}..
  v8h a = *(const v8h*)p0;               // K = k0 + 8*chunk + 0..7
  v8h b = *(const v8h*)(p0 + 16);        // K = k0 + 16 + 8*chunk + 0..7
  v16h v;
#pragma unroll
  for (int i = 0; i < 8; ++i) { v[i] = a[i]; v[8 + i] = b[i]; }
  return v;
}

// ---------------------------------------------------------------------------
// D[Mp x ldD] = A[Mp x ldA] @ B[Np x ldB]^T   (f16 in, f32 accumulate/out)
// grid = (Np/16, Mp/16, batch), block = 32 (one wave per 16x16 output tile).
// ---------------------------------------------------------------------------
__global__ void k_gemm_wmma(const _Float16* __restrict__ A,
                            const _Float16* __restrict__ B,
                            float* __restrict__ D,
                            int ldA, int ldB, int ldD, int Ktiles,
                            long strideB, long strideD) {
  int lane = threadIdx.x;
  int n0 = blockIdx.x * 16;
  int m0 = blockIdx.y * 16;
  B += (long)blockIdx.z * strideB;
  D += (long)blockIdx.z * strideD;
  v8f acc = {};
  for (int kt = 0; kt < Ktiles; ++kt) {
    v16h a = load_tile16x32(A, ldA, m0, kt * 32, lane);
    v16h b = load_tile16x32(B, ldB, n0, kt * 32, lane);
    // D = A x B + C   (v_wmma_f32_16x16x32_f16)
    acc = __builtin_amdgcn_wmma_f32_16x16x32_f16(false, a, false, b,
                                                 (short)0, acc, false, false);
  }
  // C/D layout: lane<16 -> rows m0+0..7, col n0+lane; lane>=16 -> rows m0+8..15
  int col   = n0 + (lane & 15);
  int rbase = m0 + ((lane >> 4) << 3);
#pragma unroll
  for (int r = 0; r < 8; ++r) D[(size_t)(rbase + r) * ldD + col] = acc[r];
}

// ---------------------------------------------------------------------------
// Sequential child-sum TreeLSTM scan. grid = 2 (tree), block = 512.
// IOUX[2][IOUP][128] = Wioux @ X^T (precomputed), FX[2][FXP][128] = Wfx @ X^T.
// C/H state [2][128][CHS] in global workspace (L2-resident).
// ---------------------------------------------------------------------------
__global__ void k_scan(const int* __restrict__ lch, const int* __restrict__ rch, int K,
                       const float* __restrict__ IOUX, const float* __restrict__ FX,
                       const float* __restrict__ Wiouh, const float* __restrict__ bioux,
                       const float* __restrict__ biouh,
                       const float* __restrict__ Wfh, const float* __restrict__ bfx,
                       const float* __restrict__ bfh,
                       float* __restrict__ Cw, float* __restrict__ Hw) {
  int tree = blockIdx.x;
  const int* ch = tree ? rch : lch;
  const float* ioux = IOUX + (size_t)tree * IOUP * NNODE;
  const float* fx   = FX   + (size_t)tree * FXP * NNODE;
  float* C = Cw + (size_t)tree * NNODE * CHS;
  float* H = Hw + (size_t)tree * NNODE * CHS;

  __shared__ float hsum[MEMD];
  __shared__ float iouS[IOU];
  __shared__ float fcs[MEMD];
  __shared__ int   chS[64];
  __shared__ int   nchS;

  int tid = threadIdx.x;

  for (int t = 0; t < NNODE; ++t) {
    if (tid == 0) {
      int n = 0;
      for (int k = 0; k < K; ++k) {
        int c = ch[t * K + k];
        if (c >= 0 && n < 64) chS[n++] = c;
      }
      nchS = n;
    }
    if (tid < MEMD) fcs[tid] = 0.0f;
    __syncthreads();

    int nch = nchS;
    // hsum = sum_children H[child]
    if (tid < MEMD) {
      float s = 0.0f;
      for (int k = 0; k < nch; ++k) s += H[(size_t)chS[k] * CHS + tid];
      hsum[tid] = s;
    }
    __syncthreads();

    // iou = IOUX_pre + bioux + Wiouh @ hsum + biouh
    if (tid < IOU) {
      const float* wr = Wiouh + (size_t)tid * MEMD;
      __builtin_prefetch(wr, 0, 1);
      float acc = ioux[(size_t)tid * NNODE + t] + bioux[tid] + biouh[tid];
      for (int j = 0; j < MEMD; ++j) acc += wr[j] * hsum[j];
      iouS[tid] = acc;
    }
    // forget gates: f_k = sigmoid(Wfh @ H[child_k] + bfh + FX_pre + bfx);
    // fcs += f_k * C[child_k]  (parallel over (child, row), LDS float atomics)
    for (int idx = tid; idx < nch * MEMD; idx += blockDim.x) {
      int k  = idx / MEMD;
      int jp = idx % MEMD;
      const float* hc = H + (size_t)chS[k] * CHS;
      const float* wr = Wfh + (size_t)jp * MEMD;
      float acc = fx[(size_t)jp * NNODE + t] + bfx[jp] + bfh[jp];
      for (int j = 0; j < MEMD; ++j) acc += wr[j] * hc[j];
      float f = sigf(acc);
      atomicAdd(&fcs[jp], f * C[(size_t)chS[k] * CHS + jp]);
    }
    __syncthreads();

    if (tid < MEMD) {
      float i = sigf(iouS[tid]);
      float o = sigf(iouS[MEMD + tid]);
      float u = tanhf(iouS[2 * MEMD + tid]);
      float c = i * u + fcs[tid];
      float h = o * tanhf(c);
      C[(size_t)t * CHS + tid] = c;
      H[(size_t)t * CHS + tid] = h;
    }
    __threadfence();
    __syncthreads();
  }
}

// ---------------------------------------------------------------------------
// Build conc_last = cat(emb[token[127]], C[127]) as f16 rows of [16][CONKP]
// (row 0 = left tree, row 1 = right tree, rows 2..15 zero pad).
// ---------------------------------------------------------------------------
__global__ void k_conc(const int* __restrict__ ltok, const int* __restrict__ rtok,
                       const float* __restrict__ emb, const float* __restrict__ Cw,
                       _Float16* __restrict__ conc16) {
  int idx = blockIdx.x * blockDim.x + threadIdx.x;
  if (idx >= 16 * CONKP) return;
  int row = idx / CONKP;
  int k   = idx % CONKP;
  float v = 0.0f;
  if (row < 2) {
    const int* tok = row ? rtok : ltok;
    if (k < INDIM)       v = emb[(size_t)tok[NNODE - 1] * INDIM + k];
    else if (k < CONK)   v = Cw[(size_t)row * NNODE * CHS + (size_t)(NNODE - 1) * CHS + (k - INDIM)];
  }
  conc16[idx] = (_Float16)v;
}

// ---------------------------------------------------------------------------
// GRU head (h0 = 0 => Whh contributes only bhh) + assemble lvec/rvec.
// GIf/GIb are [GH3P][16] (col = tree). vecs = [2][1050].
// ---------------------------------------------------------------------------
__global__ void k_gru(const float* __restrict__ GIf, const float* __restrict__ GIb,
                      const float* __restrict__ bih_f, const float* __restrict__ bhh_f,
                      const float* __restrict__ bih_b, const float* __restrict__ bhh_b,
                      const float* __restrict__ Cw, float* __restrict__ vecs) {
  int idx = blockIdx.x * blockDim.x + threadIdx.x;
  if (idx >= 2 * CONK) return;
  int tree = idx / CONK;
  int j    = idx % CONK;

  // forward GRU cell
  {
    float ri = GIf[(size_t)j * 16 + tree]            + bih_f[j]            + bhh_f[j];
    float zi = GIf[(size_t)(CONK + j) * 16 + tree]   + bih_f[CONK + j]     + bhh_f[CONK + j];
    float ni = GIf[(size_t)(2 * CONK + j) * 16 + tree] + bih_f[2 * CONK + j];
    float r = sigf(ri), z = sigf(zi);
    float n = tanhf(ni + r * bhh_f[2 * CONK + j]);
    vecs[(size_t)tree * 1050 + MEMD + j] = (1.0f - z) * n;
  }
  // backward GRU cell
  {
    float ri = GIb[(size_t)j * 16 + tree]            + bih_b[j]            + bhh_b[j];
    float zi = GIb[(size_t)(CONK + j) * 16 + tree]   + bih_b[CONK + j]     + bhh_b[CONK + j];
    float ni = GIb[(size_t)(2 * CONK + j) * 16 + tree] + bih_b[2 * CONK + j];
    float r = sigf(ri), z = sigf(zi);
    float n = tanhf(ni + r * bhh_b[2 * CONK + j]);
    vecs[(size_t)tree * 1050 + MEMD + CONK + j] = (1.0f - z) * n;
  }
  if (j < MEMD) {
    vecs[(size_t)tree * 1050 + j] =
        Cw[(size_t)tree * NNODE * CHS + (size_t)(NNODE - 1) * CHS + j];
  }
}

// ---------------------------------------------------------------------------
// Classifier head: v = cat(l*r, |l-r|) [2100]; out = sigmoid(whW@v+whb) [50];
// logits = wpW@out+wpb [5]; log_softmax -> d_out. One block of 256.
// ---------------------------------------------------------------------------
__global__ void k_head(const float* __restrict__ vecs,
                       const float* __restrict__ whW, const float* __restrict__ whb,
                       const float* __restrict__ wpW, const float* __restrict__ wpb,
                       float* __restrict__ out) {
  __shared__ float v[2100];
  __shared__ float hid[50];
  __shared__ float logits[5];
  int tid = threadIdx.x;
  for (int i = tid; i < 1050; i += blockDim.x) {
    float a = vecs[i], b = vecs[1050 + i];
    v[i]        = a * b;
    v[1050 + i] = fabsf(a - b);
  }
  __syncthreads();
  if (tid < 50) {
    const float* w = whW + (size_t)tid * 2100;
    float acc = whb[tid];
    for (int j = 0; j < 2100; ++j) acc += w[j] * v[j];
    hid[tid] = sigf(acc);
  }
  __syncthreads();
  if (tid < 5) {
    float acc = wpb[tid];
    for (int j = 0; j < 50; ++j) acc += wpW[tid * 50 + j] * hid[j];
    logits[tid] = acc;
  }
  __syncthreads();
  if (tid == 0) {
    float m = logits[0];
    for (int i = 1; i < 5; ++i) m = fmaxf(m, logits[i]);
    float s = 0.0f;
    for (int i = 0; i < 5; ++i) s += __expf(logits[i] - m);
    float lse = m + __logf(s);
    for (int i = 0; i < 5; ++i) out[i] = logits[i] - lse;
  }
}

// ---------------------------------------------------------------------------
extern "C" void kernel_launch(void* const* d_in, const int* in_sizes, int n_in,
                              void* d_out, int out_size, void* d_ws, size_t ws_size,
                              hipStream_t stream) {
  const int*   ltok  = (const int*)d_in[0];
  const int*   rtok  = (const int*)d_in[1];
  const int*   lch   = (const int*)d_in[2];
  const int*   rch   = (const int*)d_in[3];
  const int    K     = in_sizes[2] / NNODE;
  const float* emb   = (const float*)d_in[4];
  const float* Wioux = (const float*)d_in[5];
  const float* bioux = (const float*)d_in[6];
  const float* Wiouh = (const float*)d_in[7];
  const float* biouh = (const float*)d_in[8];
  const float* Wfx   = (const float*)d_in[9];
  const float* bfx   = (const float*)d_in[10];
  const float* Wfh   = (const float*)d_in[11];
  const float* bfh   = (const float*)d_in[12];
  const float* Wih_f = (const float*)d_in[13];
  // d_in[14] = Whh_f (unused: h0 = 0)
  const float* bih_f = (const float*)d_in[15];
  const float* bhh_f = (const float*)d_in[16];
  const float* Wih_b = (const float*)d_in[17];
  // d_in[18] = Whh_b (unused)
  const float* bih_b = (const float*)d_in[19];
  const float* bhh_b = (const float*)d_in[20];
  const float* whW   = (const float*)d_in[21];
  const float* whb   = (const float*)d_in[22];
  const float* wpW   = (const float*)d_in[23];
  const float* wpb   = (const float*)d_in[24];
  float* out = (float*)d_out;

  // ---- workspace carve-up (256B aligned) ----
  char* ws = (char*)d_ws;
  size_t off = 0;
  auto alloc = [&](size_t bytes) {
    char* p = ws + off;
    off = (off + bytes + 255) & ~(size_t)255;
    return p;
  };
  _Float16* X16     = (_Float16*)alloc((size_t)2 * NNODE * INP * 2);
  _Float16* Wioux16 = (_Float16*)alloc((size_t)IOUP * INP * 2);
  _Float16* Wfx16   = (_Float16*)alloc((size_t)FXP * INP * 2);
  _Float16* Wihf16  = (_Float16*)alloc((size_t)GH3P * CONKP * 2);
  _Float16* Wihb16  = (_Float16*)alloc((size_t)GH3P * CONKP * 2);
  _Float16* conc16  = (_Float16*)alloc((size_t)16 * CONKP * 2);
  float*    IOUX    = (float*)alloc((size_t)2 * IOUP * NNODE * 4);
  float*    FX      = (float*)alloc((size_t)2 * FXP * NNODE * 4);
  float*    Cw      = (float*)alloc((size_t)2 * NNODE * CHS * 4);
  float*    Hw      = (float*)alloc((size_t)2 * NNODE * CHS * 4);
  float*    GIf     = (float*)alloc((size_t)GH3P * 16 * 4);
  float*    GIb     = (float*)alloc((size_t)GH3P * 16 * 4);
  float*    vecs    = (float*)alloc((size_t)2 * 1050 * 4);
  (void)ws_size; (void)n_in; (void)out_size;

  // ---- phase 0: convert weights to padded f16, gather embeddings ----
  {
    int n;
    n = IOUP * INP;
    k_cvt_pad<<<(n + 255) / 256, 256, 0, stream>>>(Wioux, Wioux16, IOU, INDIM, IOUP, INP);
    n = FXP * INP;
    k_cvt_pad<<<(n + 255) / 256, 256, 0, stream>>>(Wfx, Wfx16, MEMD, INDIM, FXP, INP);
    n = GH3P * CONKP;
    k_cvt_pad<<<(n + 255) / 256, 256, 0, stream>>>(Wih_f, Wihf16, GH3, CONK, GH3P, CONKP);
    k_cvt_pad<<<(n + 255) / 256, 256, 0, stream>>>(Wih_b, Wihb16, GH3, CONK, GH3P, CONKP);
    n = 2 * NNODE * INP;
    k_gather<<<(n + 255) / 256, 256, 0, stream>>>(ltok, rtok, emb, X16);
  }

  // ---- phase 1: batched WMMA GEMMs, both trees ----
  // IOUX[tree] = Wioux @ X[tree]^T     (464x320 @ 320x128)
  k_gemm_wmma<<<dim3(NNODE / 16, IOUP / 16, 2), 32, 0, stream>>>(
      Wioux16, X16, IOUX, INP, INP, NNODE, INP / 32,
      (long)NNODE * INP, (long)IOUP * NNODE);
  // FX[tree] = Wfx @ X[tree]^T         (160x320 @ 320x128)
  k_gemm_wmma<<<dim3(NNODE / 16, FXP / 16, 2), 32, 0, stream>>>(
      Wfx16, X16, FX, INP, INP, NNODE, INP / 32,
      (long)NNODE * INP, (long)FXP * NNODE);

  // ---- phase 2: sequential TreeLSTM scan (1 workgroup per tree) ----
  k_scan<<<2, 512, 0, stream>>>(lch, rch, K, IOUX, FX, Wiouh, bioux, biouh,
                                Wfh, bfx, bfh, Cw, Hw);

  // ---- phase 3: GRU input projections via WMMA ----
  {
    int n = 16 * CONKP;
    k_conc<<<(n + 255) / 256, 256, 0, stream>>>(ltok, rtok, emb, Cw, conc16);
  }
  k_gemm_wmma<<<dim3(1, GH3P / 16, 1), 32, 0, stream>>>(
      Wihf16, conc16, GIf, CONKP, CONKP, 16, CONKP / 32, 0L, 0L);
  k_gemm_wmma<<<dim3(1, GH3P / 16, 1), 32, 0, stream>>>(
      Wihb16, conc16, GIb, CONKP, CONKP, 16, CONKP / 32, 0L, 0L);

  // ---- phase 4: GRU gates + vector assembly, then classifier head ----
  {
    int n = 2 * CONK;
    k_gru<<<(n + 255) / 256, 256, 0, stream>>>(GIf, GIb, bih_f, bhh_f, bih_b, bhh_b,
                                               Cw, vecs);
  }
  k_head<<<1, 256, 0, stream>>>(vecs, whW, whb, wpW, wpb, out);
}
